// BERT_BiLSTM_CRF_62972810494585
// MI455X (gfx1250) — matrix operand order repaired and checked by reading, not compiled, not run on hardware
//
#include <hip/hip_runtime.h>

// CRF NLL on MI455X (gfx1250):
//  - forward algorithm recast as exp-space 32x32x32 matmuls -> v_wmma_f32_16x16x32_f16
//  - S split into 16 chunks/batch (8192 waves) + log-semiring tree combine for parallelism
//  - per-column max normalization + f32 log-scale accumulators for stability
//  - rho row-permutation makes D-tile -> B-tile repack shuffle-free
//  - emit factors loaded directly as per-lane float2 (no bpermute gather in hot loop)

constexpr int T     = 32;    // TAGSET
constexpr int START = 30;
constexpr int STOP  = 31;
constexpr int B     = 512;
constexpr int S     = 2048;
constexpr int NC    = 16;    // chunks per batch
constexpr int CLEN  = S / NC;

typedef _Float16 v16h __attribute__((ext_vector_type(16)));
typedef float    v8f  __attribute__((ext_vector_type(8)));

constexpr float ASCALE_INV   = 1.0f / 64.0f;          // pre-scale exp(trans) to keep f16 products < 65504
constexpr float LOG_ASCALE   = 4.1588830833596715f;   // ln(64), re-added to the column scales
constexpr float COLMAX_GUARD = 1e-30f;

static __device__ __forceinline__ v8f wmma16(v16h a, v16h b, v8f c) {
  return __builtin_amdgcn_wmma_f32_16x16x32_f16(false, a, false, b, (short)0, c, false, false);
}

static __device__ __forceinline__ float wmax8(v8f a) {
  return fmaxf(fmaxf(fmaxf(a[0], a[1]), fmaxf(a[2], a[3])),
               fmaxf(fmaxf(a[4], a[5]), fmaxf(a[6], a[7])));
}

static __device__ __forceinline__ v16h splat_h16(float x) {
  _Float16 hx = (_Float16)x;
  v16h v;
#pragma unroll
  for (int i = 0; i < 16; ++i) v[i] = hx;
  return v;
}

// K slot in this code's contiguous f16 A order (k = c + 16h, matching B) for packed component c (=2v+p), lane half h
static __device__ __forceinline__ int a_kslot(int c, int h) {
  int v = c >> 1, p = c & 1;
  return 2 * v + p + 16 * h;
}

// ---------------------------------------------------------------------------
// Kernel 1: per-(batch,chunk) exp-space product matrix of 128 elementary steps
// ---------------------------------------------------------------------------
__global__ __launch_bounds__(256) void crf_chunk_scan(
    const float* __restrict__ em, const float* __restrict__ trans,
    float* __restrict__ wsQ, float* __restrict__ wsC)
{
  const int wid   = (blockIdx.x * blockDim.x + threadIdx.x) >> 5;
  const int lane  = threadIdx.x & 31;
  const int b     = wid >> 4;          // / NC
  const int chunk = wid & (NC - 1);
  if (b >= B) return;
  const int h  = lane >> 4;
  const int lq = lane & 15;

  // Constant A tiles: exp(trans)/64 in WMMA A-layout; A-row slot M holds logical
  // tag rho(M) = 2*(M%8) + (M/16) + 16*((M/8)%2) so D tiles repack into B tiles
  // with pure per-lane ops.
  v16h Ah[2];
#pragma unroll
  for (int m = 0; m < 2; ++m) {
    const int row = 2 * (lq & 7) + m + 16 * (lq >> 3);   // rho(16*m + lq)
#pragma unroll
    for (int v = 0; v < 8; ++v) {
      const int k0 = a_kslot(2 * v, h);
      const float2 tr = *(const float2*)(trans + row * T + k0);
      Ah[m][2 * v]     = (_Float16)(__expf(tr.x) * ASCALE_INV);
      Ah[m][2 * v + 1] = (_Float16)(__expf(tr.y) * ASCALE_INV);
    }
  }

  // Running chunk product Q = identity (exp-space), per-column scales = 0.
  v16h qB[2];
  float cacc0 = 0.f, cacc1 = 0.f;
#pragma unroll
  for (int n = 0; n < 2; ++n) {
    const int j = lq + 16 * n;
#pragma unroll
    for (int c = 0; c < 16; ++c) {
      const int K = c + 16 * h;            // B-layout: K = 2v+p+16h
      qB[n][c] = (_Float16)((K == j) ? 1.0f : 0.0f);
    }
  }

  const int tstart = chunk << 7;           // chunk * 128 (low bits provably 0)
  const int tmax   = tstart + CLEN - 1;
  // Each lane directly loads the two emit values its A-tile rows need:
  // tags ta0 = rho(16*0 + lq) and ta0+1 = rho(16*1 + lq); ta0 is even -> float2.
  const int ta0 = 2 * (lq & 7) + 16 * (lq >> 3);
  const float* ebase = em + (size_t)b * S * T + ta0;

  float2 ebuf[8];                          // 8-deep load ring hides HBM latency
#pragma unroll
  for (int i = 0; i < 8; ++i)
    ebuf[i] = *(const float2*)(ebase + (size_t)(tstart + i) * T);

  const v8f zero = {0.f, 0.f, 0.f, 0.f, 0.f, 0.f, 0.f, 0.f};

  for (int tb = 0; tb < CLEN; tb += 8) {
#pragma unroll
    for (int u = 0; u < 8; ++u) {
      const int t = tstart + tb + u;
      int tn = t + 8; if (tn > tmax) tn = tmax;
      const float2 enext = *(const float2*)(ebase + (size_t)tn * T);

      const float em0 = __expf(ebuf[u].x);          // exp(emit) for A-tile 0 row
      const float em1 = __expf(ebuf[u].y);          // exp(emit) for A-tile 1 row
      const v16h A0 = Ah[0] * splat_h16(em0);       // fold emit into A rows (pk_mul_f16)
      const v16h A1 = Ah[1] * splat_h16(em1);

      const v8f D00 = wmma16(A0, qB[0], zero);      // R = diag(e)·exp(trans) @ Q
      const v8f D10 = wmma16(A1, qB[0], zero);
      const v8f D01 = wmma16(A0, qB[1], zero);
      const v8f D11 = wmma16(A1, qB[1], zero);

      // two-phase column max: local trees, then both cross-half swaps batched
      const float l0 = fmaxf(wmax8(D00), wmax8(D10));
      const float l1 = fmaxf(wmax8(D01), wmax8(D11));
      const float s0 = __shfl_xor(l0, 16, 32);
      const float s1 = __shfl_xor(l1, 16, 32);
      const float m0 = fmaxf(fmaxf(l0, s0), COLMAX_GUARD);
      const float m1 = fmaxf(fmaxf(l1, s1), COLMAX_GUARD);
      const float r0 = __builtin_amdgcn_rcpf(m0);
      const float r1 = __builtin_amdgcn_rcpf(m1);
      cacc0 += __logf(m0) + LOG_ASCALE;
      cacc1 += __logf(m1) + LOG_ASCALE;

#pragma unroll
      for (int v = 0; v < 8; ++v) {                 // shuffle-free D -> B repack
        qB[0][2 * v]     = (_Float16)(D00[v] * r0);
        qB[0][2 * v + 1] = (_Float16)(D10[v] * r0);
        qB[1][2 * v]     = (_Float16)(D01[v] * r1);
        qB[1][2 * v + 1] = (_Float16)(D11[v] * r1);
      }
      ebuf[u] = enext;
    }
  }

  // Store chunk result: Q row-major f32 [k][j], per-column scales c[j].
  const size_t qoff = ((size_t)b * NC + chunk) * (T * T);
  const size_t coff = ((size_t)b * NC + chunk) * T;
#pragma unroll
  for (int c = 0; c < 16; ++c) {
    const int K = c + 16 * h;
    wsQ[qoff + K * T + lq]      = (float)qB[0][c];
    wsQ[qoff + K * T + lq + 16] = (float)qB[1][c];
  }
  if (h == 0) {
    wsC[coff + lq]      = cacc0;
    wsC[coff + lq + 16] = cacc1;
  }
}

// ---------------------------------------------------------------------------
// Kernel 2: binary-tree log-semiring combine of chunk matrices (in place)
// ---------------------------------------------------------------------------
__global__ __launch_bounds__(256) void crf_combine(
    float* __restrict__ wsQ, float* __restrict__ wsC, int stride, int pairs)
{
  const int wid  = (blockIdx.x * blockDim.x + threadIdx.x) >> 5;
  const int lane = threadIdx.x & 31;
  const int b    = wid / pairs;
  const int p    = wid % pairs;
  if (b >= B) return;
  const int h  = lane >> 4;
  const int lq = lane & 15;

  const int sa = p * (stride * 2);           // earlier chunk (right operand)
  const int sb = sa + stride;                // later chunk   (left operand)
  const size_t qa = ((size_t)b * NC + sa) * (T * T);
  const size_t qb = ((size_t)b * NC + sb) * (T * T);
  const size_t ca = ((size_t)b * NC + sa) * T;
  const size_t cb = ((size_t)b * NC + sb) * T;

  // later chunk's column scales become bounded row weights
  const float cbl = wsC[cb + lane];
  float mb = cbl;
#pragma unroll
  for (int o = 16; o >= 1; o >>= 1) mb = fmaxf(mb, __shfl_xor(mb, o, 32));
  const float w = __expf(cbl - mb);

  // A operand: later Q in A-layout (rho rows), f16
  v16h Aop[2];
#pragma unroll
  for (int m = 0; m < 2; ++m) {
    const int row = 2 * (lq & 7) + m + 16 * (lq >> 3);
#pragma unroll
    for (int v = 0; v < 8; ++v) {
      const int k0 = a_kslot(2 * v, h);
      const float2 q2 = *(const float2*)(wsQ + qb + row * T + k0);
      Aop[m][2 * v]     = (_Float16)q2.x;
      Aop[m][2 * v + 1] = (_Float16)q2.y;
    }
  }

  // B operand: earlier Q row-scaled by w, f16 B-layout
  v16h Bop[2];
#pragma unroll
  for (int c = 0; c < 16; ++c) {
    const int K = c + 16 * h;
    const float wk = __shfl(w, K, 32);
    Bop[0][c] = (_Float16)(wsQ[qa + K * T + lq]      * wk);
    Bop[1][c] = (_Float16)(wsQ[qa + K * T + lq + 16] * wk);
  }

  const float ca0 = wsC[ca + lq];
  const float ca1 = wsC[ca + lq + 16];

  const v8f zero = {0.f, 0.f, 0.f, 0.f, 0.f, 0.f, 0.f, 0.f};
  const v8f D00 = wmma16(Aop[0], Bop[0], zero);
  const v8f D10 = wmma16(Aop[1], Bop[0], zero);
  const v8f D01 = wmma16(Aop[0], Bop[1], zero);
  const v8f D11 = wmma16(Aop[1], Bop[1], zero);

  const float l0 = fmaxf(wmax8(D00), wmax8(D10));
  const float l1 = fmaxf(wmax8(D01), wmax8(D11));
  const float s0 = __shfl_xor(l0, 16, 32);
  const float s1 = __shfl_xor(l1, 16, 32);
  const float m0 = fmaxf(fmaxf(l0, s0), COLMAX_GUARD);
  const float m1 = fmaxf(fmaxf(l1, s1), COLMAX_GUARD);
  const float r0 = __builtin_amdgcn_rcpf(m0);
  const float r1 = __builtin_amdgcn_rcpf(m1);

#pragma unroll
  for (int v = 0; v < 8; ++v) {              // D slot (m,v,h) = logical row 2v+m+16h
    wsQ[qa + (2 * v     + 16 * h) * T + lq]      = D00[v] * r0;
    wsQ[qa + (2 * v + 1 + 16 * h) * T + lq]      = D10[v] * r0;
    wsQ[qa + (2 * v     + 16 * h) * T + lq + 16] = D01[v] * r1;
    wsQ[qa + (2 * v + 1 + 16 * h) * T + lq + 16] = D11[v] * r1;
  }
  if (h == 0) {
    wsC[ca + lq]      = ca0 + mb + __logf(m0);
    wsC[ca + lq + 16] = ca1 + mb + __logf(m1);
  }
}

// ---------------------------------------------------------------------------
// Kernel 3: logZ per batch. Initial vector is the START delta, so
// alpha_final[i] = c[START] + log Q[i, START].
// ---------------------------------------------------------------------------
__global__ __launch_bounds__(256) void crf_finalize(
    const float* __restrict__ trans,
    const float* __restrict__ wsQ, const float* __restrict__ wsC,
    float* __restrict__ logZ)
{
  const int wid  = (blockIdx.x * blockDim.x + threadIdx.x) >> 5;
  const int lane = threadIdx.x & 31;
  if (wid >= B) return;
  const size_t q0 = (size_t)wid * NC * (T * T);
  const size_t c0 = (size_t)wid * NC * T;

  const float cS = wsC[c0 + START];
  const float qi = wsQ[q0 + lane * T + START];
  const float endv = cS + __logf(qi) + trans[STOP * T + lane];

  float wm = endv;
#pragma unroll
  for (int o = 16; o >= 1; o >>= 1) wm = fmaxf(wm, __shfl_xor(wm, o, 32));
  float sum = __expf(endv - wm);
#pragma unroll
  for (int o = 16; o >= 1; o >>= 1) sum += __shfl_xor(sum, o, 32);
  if (lane == 0) logZ[wid] = wm + __logf(sum);
}

// ---------------------------------------------------------------------------
// Kernel 4: gold path scores (gather-reduce), deterministic block reduction
// ---------------------------------------------------------------------------
__global__ __launch_bounds__(256) void crf_gold(
    const float* __restrict__ em, const float* __restrict__ trans,
    const int* __restrict__ tags, float* __restrict__ gold)
{
  const int b   = blockIdx.x;
  const int tid = threadIdx.x;
  const int* tg = tags + (size_t)b * S;
  float loc = 0.f;
  for (int t = tid; t < S; t += 256) {
    const int ct = tg[t];
    loc += em[((size_t)b * S + t) * T + ct];
    const int pv = (t == 0) ? START : tg[t - 1];
    loc += trans[ct * T + pv];
    if (t == S - 1) loc += trans[STOP * T + ct];
  }
  __shared__ float red[256];
  red[tid] = loc;
  __syncthreads();
#pragma unroll
  for (int s = 128; s > 0; s >>= 1) {
    if (tid < s) red[tid] += red[tid + s];
    __syncthreads();
  }
  if (tid == 0) gold[b] = red[0];
}

// ---------------------------------------------------------------------------
// Kernel 5: final scalar = sum_b (logZ - gold)
// ---------------------------------------------------------------------------
__global__ __launch_bounds__(512) void crf_reduce(
    const float* __restrict__ logZ, const float* __restrict__ gold,
    float* __restrict__ out)
{
  const int tid = threadIdx.x;
  __shared__ float red[512];
  red[tid] = logZ[tid] - gold[tid];
  __syncthreads();
#pragma unroll
  for (int s = 256; s > 0; s >>= 1) {
    if (tid < s) red[tid] += red[tid + s];
    __syncthreads();
  }
  if (tid == 0) out[0] = red[0];
}

extern "C" void kernel_launch(void* const* d_in, const int* in_sizes, int n_in,
                              void* d_out, int out_size, void* d_ws, size_t ws_size,
                              hipStream_t stream)
{
  (void)in_sizes; (void)n_in; (void)out_size; (void)ws_size;
  const float* em    = (const float*)d_in[0];
  const float* trans = (const float*)d_in[1];
  const int*   tags  = (const int*)d_in[2];
  float* out = (float*)d_out;

  float* wsQ  = (float*)d_ws;                      // B*NC*1024 f32 (~33.5 MB)
  float* wsC  = wsQ + (size_t)B * NC * (T * T);    // B*NC*32 f32
  float* logZ = wsC + (size_t)B * NC * T;          // B f32
  float* gold = logZ + B;                          // B f32

  // 1) 8192 waves build chunk product matrices with WMMA
  crf_chunk_scan<<<(B * NC) / 8, 256, 0, stream>>>(em, trans, wsQ, wsC);

  // 2) 4-level tree combine across chunks (WMMA log-semiring matmuls)
  for (int s = 1; s < NC; s <<= 1) {
    const int pairs  = NC / (2 * s);
    const int waves  = B * pairs;
    const int blocks = (waves * 32 + 255) / 256;
    crf_combine<<<blocks, 256, 0, stream>>>(wsQ, wsC, s, pairs);
  }

  // 3) log partition per batch
  crf_finalize<<<(B * 32 + 255) / 256, 256, 0, stream>>>(trans, wsQ, wsC, logZ);

  // 4) gold scores
  crf_gold<<<B, 256, 0, stream>>>(em, trans, tags, gold);

  // 5) final scalar
  crf_reduce<<<1, 512, 0, stream>>>(logZ, gold, out);
}
